// MultiHeadAttention_30657476559110
// MI455X (gfx1250) — compile-verified, matrix operand order checked by reading, not run on hardware
//
#include <hip/hip_runtime.h>
#include <hip/hip_bf16.h>
#include <stdint.h>

// Problem constants (match reference)
#define B_ 8
#define S_ 1024
#define F_ 256
#define H_ 8
#define HF_ (H_ * F_)      // 2048
#define M_ (B_ * S_)       // 8192
#define NEGV (-1e18f)

typedef __attribute__((ext_vector_type(16))) _Float16 v16h;
typedef __attribute__((ext_vector_type(8)))  _Float16 v8h;
typedef __attribute__((ext_vector_type(8)))  float    v8f;
typedef __attribute__((ext_vector_type(4)))  unsigned int u32x4;
typedef __attribute__((ext_vector_type(8)))  int          i32x8;
typedef __attribute__((ext_vector_type(4)))  int          i32x4;

// ---------------------------------------------------------------------------
// 16x32 f16 fragment load (A-matrix layout; also used for B since CDNA5 B
// fragments for col n use the same K-packing as A fragments for row m).
// Lane 0-15 : row = lane,    K = {0..7, 16..23}
// Lane 16-31: row = lane-16, K = {8..15, 24..31}
// ---------------------------------------------------------------------------
__device__ __forceinline__ v16h frag16x32(const _Float16* __restrict__ base, int ld) {
  const int lane = threadIdx.x & 31;
  const int row  = lane & 15;
  const int kb   = (lane >> 4) << 3;            // 0 or 8
  const _Float16* p = base + (size_t)row * ld + kb;
  v8h lo = *(const v8h*)p;                      // K = kb .. kb+7
  v8h hi = *(const v8h*)(p + 16);               // K = kb+16 .. kb+23
  v16h a;
#pragma unroll
  for (int i = 0; i < 8; ++i) { a[i] = lo[i]; a[i + 8] = hi[i]; }
  return a;
}

__device__ __forceinline__ v8f wmma32(v16h a, v16h b, v8f c) {
  // D = A(16x32 f16) x B(32x16 f16) + C(16x16 f32)
  return __builtin_amdgcn_wmma_f32_16x16x32_f16(
      false, a, false, b, (short)0, c, false, false);
}

// ---------------------------------------------------------------------------
// Tensor Data Mover: 2-D tile load Global -> LDS (ISA 08_async_tensor §8).
// D# group0: count=1 | lds_addr | global_addr[56:0] | type=2
// D# group1: data_size, tensor_dim0/1, tile_dim0/1, tensor_dim0_stride
// Groups 2/3 (dims 3/4) zero = unused (tensor <= 2D). Issued once per calling
// wave; completion via TENSORcnt (s_wait_tensorcnt).
// This toolchain's builtin takes 6 args (g0, g1, g2, g3, extra, cpol).
// ---------------------------------------------------------------------------
__device__ __forceinline__ void tdm_load_2d(uint32_t lds_off, const void* gptr,
                                            uint32_t dsz_code,      // 0=1B,1=2B,2=4B,3=8B
                                            uint32_t tile_d0, uint32_t tile_d1,
                                            uint32_t tensor_d0, uint32_t tensor_d1,
                                            uint64_t stride0_elems) {
  const uint64_t ga = (uint64_t)gptr;
  u32x4 g0;
  g0[0] = 1u;                                        // count=1, user D#
  g0[1] = lds_off;                                   // LDS byte address
  g0[2] = (uint32_t)ga;                              // global_addr[31:0]
  g0[3] = (uint32_t)((ga >> 32) & 0x1FFFFFFu) | (2u << 30);  // [56:32] | type=2
  i32x8 g1;
  g1[0] = (int)(dsz_code << 16);                     // wg_mask=0, no pad/iter
  g1[1] = (int)((tensor_d0 & 0xFFFFu) << 16);        // abar=0 | td0[15:0]
  g1[2] = (int)(((tensor_d0 >> 16) & 0xFFFFu) | ((tensor_d1 & 0xFFFFu) << 16));
  g1[3] = (int)(((tensor_d1 >> 16) & 0xFFFFu) | ((tile_d0 & 0xFFFFu) << 16));
  g1[4] = (int)(tile_d1 & 0xFFFFu);                  // tile_dim1 | tile_dim2=0
  g1[5] = (int)(uint32_t)(stride0_elems & 0xFFFFFFFFu);
  g1[6] = (int)(uint32_t)((stride0_elems >> 32) & 0xFFFFu);  // | stride1=0
  g1[7] = 0;
  const i32x4 z4 = {0, 0, 0, 0};
  const i32x8 z8 = {0, 0, 0, 0, 0, 0, 0, 0};
  __builtin_amdgcn_tensor_load_to_lds(g0, g1, z4, z4, z8, 0);
}

// ---------------------------------------------------------------------------
// Elementwise fp32 -> f16
// ---------------------------------------------------------------------------
__global__ void cvt_f16_kernel(const float* __restrict__ src,
                               _Float16* __restrict__ dst, int n) {
  for (int i = blockIdx.x * blockDim.x + threadIdx.x; i < n;
       i += gridDim.x * blockDim.x)
    dst[i] = (_Float16)src[i];
}

// Wt[n*K + k] = (f16) W[k*N + n]   (transpose + convert; contiguous writes)
__global__ void tcvt_f16_kernel(const float* __restrict__ W,
                                _Float16* __restrict__ Wt, int K, int N) {
  int total = K * N;
  for (int t = blockIdx.x * blockDim.x + threadIdx.x; t < total;
       t += gridDim.x * blockDim.x) {
    int k = t % K;
    int n = t / K;
    Wt[t] = (_Float16)W[(size_t)k * N + n];
  }
}

// ---------------------------------------------------------------------------
// Projection GEMM:  Y[M_,HF_] = X[M_,F_] @ W + bias, written head-major.
//   mode 0: dst[b,h,s,f]  (for q, k)      mode 1: dst[b,h,f,s]  (for v)
// One wave computes a 16(M) x 64(N) tile. Block = 8 waves.
// ---------------------------------------------------------------------------
__global__ __launch_bounds__(256) void proj_gemm_kernel(
    const _Float16* __restrict__ X,    // [M_, F_]
    const _Float16* __restrict__ Wt,   // [HF_, F_]  (transposed weights)
    const float* __restrict__ bias,    // [HF_]
    _Float16* __restrict__ dst, int mode) {
  const int lane = threadIdx.x & 31;
  const int wid  = blockIdx.x * (blockDim.x >> 5) + (threadIdx.x >> 5);
  const int nt = wid & ((HF_ / 64) - 1);          // 32 N-tiles of 64
  const int mt = wid / (HF_ / 64);                // 512 M-tiles of 16
  const int m0 = mt * 16, n0 = nt * 64;

  v8f acc[4] = {};
#pragma unroll
  for (int k = 0; k < F_; k += 32) {
    v16h a = frag16x32(X + (size_t)m0 * F_ + k, F_);
#pragma unroll
    for (int j = 0; j < 4; ++j) {
      v16h b = frag16x32(Wt + (size_t)(n0 + j * 16) * F_ + k, F_);
      acc[j] = wmma32(a, b, acc[j]);
    }
  }
  const int n   = lane & 15;
  const int mhi = (lane >> 4) << 3;
#pragma unroll
  for (int j = 0; j < 4; ++j) {
    const int N = n0 + j * 16 + n;
    const int h = N >> 8, f = N & (F_ - 1);
    const float bb = bias[N];
#pragma unroll
    for (int r = 0; r < 8; ++r) {
      const int M = m0 + r + mhi;
      const int b = M >> 10, s = M & (S_ - 1);
      const float y = acc[j][r] + bb;
      size_t idx = (mode == 1)
                       ? (((size_t)(b * H_ + h) * F_ + f) * S_ + s)
                       : (((size_t)(b * H_ + h) * S_ + s) * F_ + f);
      dst[idx] = (_Float16)y;
    }
  }
}

// ---------------------------------------------------------------------------
// Flash attention, WGP-cooperative, TDM-fed:
//   One block (8 waves) handles ONE (b,h) and a 128-row q slab (16 rows/wave).
//   Per 32-key block, wave 0 issues two Tensor-DMA loads:
//     K: contiguous 16 KB  (1-D, 8B elements)
//     V: 256 rows x 64 B strided gather, compacted into sV[f][sk_local]
//   then s_wait_tensorcnt 0; all waves share the tiles (8x less L2 traffic).
//   Online softmax; P converted C->A layout via per-wave LDS bounce.
// ---------------------------------------------------------------------------
__global__ __launch_bounds__(256) void attn_kernel(
    const _Float16* __restrict__ qp, const _Float16* __restrict__ kp,
    const _Float16* __restrict__ vt, const uint8_t* __restrict__ mask,
    _Float16* __restrict__ ctx) {
  __shared__ __align__(16) _Float16 sK[32 * F_];      // 16 KB: [sk_local][f]
  __shared__ __align__(16) _Float16 sV[F_ * 32];      // 16 KB: [f][sk_local]
  __shared__ __align__(16) _Float16 sP[8][16 * 32];   //  8 KB: per-wave P

  const int wave = threadIdx.x >> 5;
  const int lane = threadIdx.x & 31;
  const int bh   = blockIdx.x >> 3;           // 64 (b,h) pairs
  const int qblk = blockIdx.x & 7;            // 8 slabs of 128 q rows
  const int q0   = qblk * 128 + wave * 16;
  const int b    = bh >> 3;
  const int h    = bh & 7;
  const float scale = 0.0625f;                // 1/sqrt(F_)

  const _Float16* Qb = qp + ((size_t)bh * S_ + q0) * F_;
  const _Float16* Kb = kp + (size_t)bh * S_ * F_;
  const _Float16* Vb = vt + (size_t)bh * F_ * S_;

  // LDS aperture occupies the high 32 bits of a generic pointer; low 32 bits
  // are the LDS byte offset the TDM D# wants.
  const uint32_t ldsK = (uint32_t)(uintptr_t)(void*)sK;
  const uint32_t ldsV = (uint32_t)(uintptr_t)(void*)sV;

  const int n   = lane & 15;
  const int mhi = (lane >> 4) << 3;

  float mrow[8], lrow[8];
  v8f acc[16];
#pragma unroll
  for (int r = 0; r < 8; ++r) { mrow[r] = -1e30f; lrow[r] = 0.f; }
#pragma unroll
  for (int f = 0; f < 16; ++f) acc[f] = (v8f){};

  for (int s0 = 0; s0 < S_; s0 += 32) {
    __syncthreads();   // all waves' LDS reads of previous tiles complete

    // Warm L2 for the NEXT key block (global_prefetch_b8).
    if (s0 + 32 < S_) {
      __builtin_prefetch(Kb + (size_t)(s0 + 32) * F_ + threadIdx.x * 32, 0, 1);
      __builtin_prefetch(Vb + (size_t)threadIdx.x * S_ + s0 + 32, 0, 1);
    }

    if (wave == 0) {
      // K block: contiguous 16 KB = 2048 x 8B elements, 1-D tile.
      tdm_load_2d(ldsK, Kb + (size_t)s0 * F_,
                  /*dsz=*/3, /*tile_d0=*/2048, /*tile_d1=*/1,
                  /*tensor_d0=*/2048, /*tensor_d1=*/1, /*stride0=*/2048);
      // V block: 256 rows x 32 f16, row stride S_ elements -> compacted.
      tdm_load_2d(ldsV, Vb + s0,
                  /*dsz=*/1, /*tile_d0=*/32, /*tile_d1=*/256,
                  /*tensor_d0=*/1024, /*tensor_d1=*/256, /*stride0=*/S_);
      __builtin_amdgcn_s_wait_tensorcnt(0);
    }
    __syncthreads();   // TDM-written tiles visible to all waves

    // ---- scores: two 16x16 tiles over K=F_ (8 WMMA k-steps x 2) ----
    v8f c0 = {}, c1 = {};
#pragma unroll
    for (int k = 0; k < F_; k += 32) {
      v16h aq = frag16x32(Qb + k, F_);              // from global (L2-hot)
      v16h b0 = frag16x32(sK + k, F_);              // sk 0..15  (LDS)
      v16h b1 = frag16x32(sK + 16 * F_ + k, F_);    // sk 16..31 (LDS)
      c0 = wmma32(aq, b0, c0);
      c1 = wmma32(aq, b1, c1);
    }
    const float msk0 = mask[b * S_ + s0 + n]      ? NEGV : 0.f;
    const float msk1 = mask[b * S_ + s0 + 16 + n] ? NEGV : 0.f;

    // ---- online softmax (row reductions across 16-lane halves) ----
    float corr[8];
#pragma unroll
    for (int r = 0; r < 8; ++r) {
      float s0v = c0[r] * scale + msk0;
      float s1v = c1[r] * scale + msk1;
      float mx = fmaxf(s0v, s1v);
#pragma unroll
      for (int off = 1; off < 16; off <<= 1)
        mx = fmaxf(mx, __shfl_xor(mx, off, 16));
      const float mnew = fmaxf(mrow[r], mx);
      corr[r] = __expf(mrow[r] - mnew);
      mrow[r] = mnew;
      const float p0 = __expf(s0v - mnew);
      const float p1 = __expf(s1v - mnew);
      c0[r] = p0; c1[r] = p1;
      float rs = p0 + p1;
#pragma unroll
      for (int off = 1; off < 16; off <<= 1)
        rs += __shfl_xor(rs, off, 16);
      lrow[r] = lrow[r] * corr[r] + rs;
    }
#pragma unroll
    for (int f = 0; f < 16; ++f)
#pragma unroll
      for (int r = 0; r < 8; ++r) acc[f][r] *= corr[r];

    // ---- P: C-layout -> 16x32 row-major in LDS -> A fragment ----
#pragma unroll
    for (int r = 0; r < 8; ++r) {
      sP[wave][(r + mhi) * 32 + n]      = (_Float16)c0[r];
      sP[wave][(r + mhi) * 32 + n + 16] = (_Float16)c1[r];
    }
    v16h ap = frag16x32(&sP[wave][0], 32);   // same-wave DS: in-order

    // ---- O += P(16x32) @ V(32x256), 16 f-chunks from LDS ----
#pragma unroll
    for (int f = 0; f < 16; ++f) {
      v16h bv = frag16x32(sV + (f * 16) * 32, 32);
      acc[f] = wmma32(ap, bv, acc[f]);
    }
  }

  // ---- normalize and write ctx[b, q0+m, h*F + f*16 + n] ----
#pragma unroll
  for (int f = 0; f < 16; ++f) {
#pragma unroll
    for (int r = 0; r < 8; ++r) {
      const float o = acc[f][r] / lrow[r];
      const int m = r + mhi;
      ctx[((size_t)b * S_ + q0 + m) * HF_ + h * F_ + f * 16 + n] = (_Float16)o;
    }
  }
}

// ---------------------------------------------------------------------------
// Output GEMM: out[M_,F_] = ctx[M_,HF_] @ Wo + bo   (fp32 out)
// ---------------------------------------------------------------------------
__global__ __launch_bounds__(256) void out_gemm_kernel(
    const _Float16* __restrict__ ctx,  // [M_, HF_]
    const _Float16* __restrict__ Wot,  // [F_, HF_] (transposed Wo)
    const float* __restrict__ bo,      // [F_]
    float* __restrict__ out) {         // [M_, F_]
  const int lane = threadIdx.x & 31;
  const int wid  = blockIdx.x * (blockDim.x >> 5) + (threadIdx.x >> 5);
  const int nt = wid & 3;                    // 4 N-tiles of 64
  const int mt = wid >> 2;                   // 512 M-tiles of 16
  const int m0 = mt * 16, n0 = nt * 64;

  v8f acc[4] = {};
  for (int k = 0; k < HF_; k += 32) {
    v16h a = frag16x32(ctx + (size_t)m0 * HF_ + k, HF_);
#pragma unroll
    for (int j = 0; j < 4; ++j) {
      v16h b = frag16x32(Wot + (size_t)(n0 + j * 16) * HF_ + k, HF_);
      acc[j] = wmma32(a, b, acc[j]);
    }
  }
  const int n   = lane & 15;
  const int mhi = (lane >> 4) << 3;
#pragma unroll
  for (int j = 0; j < 4; ++j) {
    const int N = n0 + j * 16 + n;
    const float bb = bo[N];
#pragma unroll
    for (int r = 0; r < 8; ++r) {
      const int M = m0 + r + mhi;
      out[(size_t)M * F_ + N] = acc[j][r] + bb;
    }
  }
}

// ---------------------------------------------------------------------------
extern "C" void kernel_launch(void* const* d_in, const int* in_sizes, int n_in,
                              void* d_out, int out_size, void* d_ws,
                              size_t ws_size, hipStream_t stream) {
  (void)in_sizes; (void)n_in; (void)out_size; (void)ws_size;

  const float* Q    = (const float*)d_in[0];
  const float* K    = (const float*)d_in[1];
  const float* V    = (const float*)d_in[2];
  const uint8_t* mk = (const uint8_t*)d_in[3];
  const float* Wq = (const float*)d_in[4];  const float* bq = (const float*)d_in[5];
  const float* Wk = (const float*)d_in[6];  const float* bk = (const float*)d_in[7];
  const float* Wv = (const float*)d_in[8];  const float* bv = (const float*)d_in[9];
  const float* Wo = (const float*)d_in[10]; const float* bo = (const float*)d_in[11];
  float* out = (float*)d_out;

  size_t off = 0;
  auto carve = [&](size_t bytes) {
    void* p = (char*)d_ws + off;
    off += (bytes + 255) & ~(size_t)255;
    return p;
  };
  _Float16* Xq  = (_Float16*)carve((size_t)M_ * F_ * 2);
  _Float16* Xk  = (_Float16*)carve((size_t)M_ * F_ * 2);
  _Float16* Xv  = (_Float16*)carve((size_t)M_ * F_ * 2);
  _Float16* Wqt = (_Float16*)carve((size_t)HF_ * F_ * 2);
  _Float16* Wkt = (_Float16*)carve((size_t)HF_ * F_ * 2);
  _Float16* Wvt = (_Float16*)carve((size_t)HF_ * F_ * 2);
  _Float16* Wot = (_Float16*)carve((size_t)F_ * HF_ * 2);
  _Float16* qp  = (_Float16*)carve((size_t)B_ * H_ * S_ * F_ * 2);
  _Float16* kp  = (_Float16*)carve((size_t)B_ * H_ * S_ * F_ * 2);
  _Float16* vtb = (_Float16*)carve((size_t)B_ * H_ * F_ * S_ * 2);
  _Float16* ctx = (_Float16*)carve((size_t)M_ * HF_ * 2);

  const int nx = M_ * F_;
  cvt_f16_kernel<<<1024, 256, 0, stream>>>(Q, Xq, nx);
  cvt_f16_kernel<<<1024, 256, 0, stream>>>(K, Xk, nx);
  cvt_f16_kernel<<<1024, 256, 0, stream>>>(V, Xv, nx);
  tcvt_f16_kernel<<<1024, 256, 0, stream>>>(Wq, Wqt, F_, HF_);
  tcvt_f16_kernel<<<1024, 256, 0, stream>>>(Wk, Wkt, F_, HF_);
  tcvt_f16_kernel<<<1024, 256, 0, stream>>>(Wv, Wvt, F_, HF_);
  tcvt_f16_kernel<<<1024, 256, 0, stream>>>(Wo, Wot, HF_, F_);

  proj_gemm_kernel<<<2048, 256, 0, stream>>>(Xq, Wqt, bq, qp, 0);
  proj_gemm_kernel<<<2048, 256, 0, stream>>>(Xk, Wkt, bk, kp, 0);
  proj_gemm_kernel<<<2048, 256, 0, stream>>>(Xv, Wvt, bv, vtb, 1);

  attn_kernel<<<512, 256, 0, stream>>>(qp, kp, vtb, mk, ctx);

  out_gemm_kernel<<<256, 256, 0, stream>>>(ctx, Wot, bo, out);
}
